// Attention_45277545234682
// MI455X (gfx1250) — compile-verified, matrix operand order checked by reading, not run on hardware
//
#include <hip/hip_runtime.h>

// ---------------------------------------------------------------------------
// MI455X (gfx1250) fused multi-head attention, f16 WMMA + flash softmax.
// wave32; WMMA 16x16x32 f16->f32. Row-sum via ones-column WMMA; row-max via
// DPP16 butterfly. K/V chunks staged once per block into LDS (TDM async
// tensor loads when available, double-buffered behind s_wait_tensorcnt).
// ---------------------------------------------------------------------------

typedef __attribute__((ext_vector_type(16))) _Float16 v16h;
typedef __attribute__((ext_vector_type(8)))  _Float16 v8h;
typedef __attribute__((ext_vector_type(4)))  _Float16 v4h;
typedef __attribute__((ext_vector_type(8)))  float    v8f;
typedef __attribute__((ext_vector_type(4)))  unsigned int u32x4;
typedef __attribute__((ext_vector_type(8)))  int i32x8;
typedef __attribute__((ext_vector_type(4)))  int i32x4;

union V16 { v16h v; v8h h[2]; };

constexpr int B = 16, L = 1024, S = 1024, D = 512, P = 512, H = 8, E = 64;
constexpr int MROWS = B * L;   // 16384 rows for the projection GEMMs

#if __has_builtin(__builtin_amdgcn_tensor_load_to_lds) && \
    __has_builtin(__builtin_amdgcn_s_wait_tensorcnt)
#define USE_TDM 1
#else
#define USE_TDM 0
#endif

// 16-lane max butterfly on the VALU: quad_perm(xor1), quad_perm(xor2),
// row_half_mirror, row_mirror. Stays within each 16-lane C-matrix group.
__device__ __forceinline__ float rowmax16_dpp(float x) {
  int xi, t;
  xi = __float_as_int(x);
  t = __builtin_amdgcn_update_dpp(xi, xi, 0xB1, 0xF, 0xF, false);   // [1,0,3,2]
  x = fmaxf(x, __int_as_float(t));  xi = __float_as_int(x);
  t = __builtin_amdgcn_update_dpp(xi, xi, 0x4E, 0xF, 0xF, false);   // [2,3,0,1]
  x = fmaxf(x, __int_as_float(t));  xi = __float_as_int(x);
  t = __builtin_amdgcn_update_dpp(xi, xi, 0x141, 0xF, 0xF, false);  // row_half_mirror
  x = fmaxf(x, __int_as_float(t));  xi = __float_as_int(x);
  t = __builtin_amdgcn_update_dpp(xi, xi, 0x140, 0xF, 0xF, false);  // row_mirror
  x = fmaxf(x, __int_as_float(t));
  return x;
}

// Low 32 bits of a generic pointer into LDS == LDS byte address (ISA 10.2).
__device__ __forceinline__ unsigned int lds_off(const void* p) {
  return (unsigned int)(unsigned long long)(size_t)p;
}

#if USE_TDM
// 2D tile TDM load: `rows` rows of `row_elems` f16 elements, row stride
// `stride_elems`, global tile start `gptr`, contiguous into LDS at lds_addr.
// D# fields per CDNA5 ISA.
// This toolchain exposes the 6-arg builtin:
//   (u32x4 g0, i32x8 g1, i32x4 g2, i32x4 g3, i32x8 pad, i32 cpol)
__device__ __forceinline__ void tdm_load_2d(
    unsigned int lds_addr, const void* gptr,
    unsigned int row_elems, unsigned int rows, unsigned int stride_elems) {
  const unsigned long long ga = (unsigned long long)(size_t)gptr;
  u32x4 g0;
  g0[0] = 1u;                                      // count=1, user descriptor
  g0[1] = lds_addr;                                // lds_addr       [63:32]
  g0[2] = (unsigned int)(ga & 0xFFFFFFFFu);        // global_addr lo [95:64]
  g0[3] = (unsigned int)(ga >> 32) | 0x80000000u;  // addr hi + type=2 [127:126]
  i32x8 g1;
  g1[0] = (int)(1u << 16);                         // data_size=1 (2 bytes)
  g1[1] = (int)(row_elems << 16);                  // tensor_dim0[15:0] @ [63:48]
  g1[2] = (int)((row_elems >> 16) | (rows << 16)); // dim0 hi | tensor_dim1 lo
  g1[3] = (int)((rows >> 16) | (row_elems << 16)); // dim1 hi | tile_dim0
  g1[4] = (int)rows;                               // tile_dim1, tile_dim2=0
  g1[5] = (int)stride_elems;                       // tensor_dim0_stride lo
  g1[6] = 0;                                       // stride hi | dim1_stride lo
  g1[7] = 0;
  const i32x4 z4 = {0, 0, 0, 0};                   // groups 2/3 unused (2D)
  const i32x8 z8 = {0, 0, 0, 0, 0, 0, 0, 0};
  __builtin_amdgcn_tensor_load_to_lds(g0, g1, z4, z4, z8, 0);
}
#endif

// ---------------------------------------------------------------------------
// f32 -> f16 convert, 4 elements / thread (n must be a multiple of 1024)
// ---------------------------------------------------------------------------
__global__ __launch_bounds__(256) void cvt_f32_to_f16(
    const float* __restrict__ in, _Float16* __restrict__ out) {
  const int i = blockIdx.x * 256 + threadIdx.x;
  float4 f = ((const float4*)in)[i];
  v4h o;
  o[0] = (_Float16)f.x; o[1] = (_Float16)f.y;
  o[2] = (_Float16)f.z; o[3] = (_Float16)f.w;
  ((v4h*)out)[i] = o;
}

// ---------------------------------------------------------------------------
// W[512,512] f32 -> Wt[512,512] f16 with Wt[n][k] = W[k][n]
// ---------------------------------------------------------------------------
__global__ __launch_bounds__(256) void transpose_cvt_w(
    const float* __restrict__ W, _Float16* __restrict__ Wt) {
  const int i = blockIdx.x * 256 + threadIdx.x;   // over 512*512
  const int k = i >> 9;
  const int n = i & 511;
  Wt[n * 512 + k] = (_Float16)W[k * 512 + n];
}

// ---------------------------------------------------------------------------
// GEMM: X[M,512](f16, row-major) @ Wt[512,512](f16, pre-transposed) + bias
// 8 waves (4 along M x 2 along N); wave tile 32x64; block tile 128x128.
// MODE 0: f16 out, head-major  [B,H,L,64]   (for Q and K)
// MODE 1: f16 out, per-head V^T [B,H,64,S]  (for V)
// MODE 2: f32 out, row-major [M,512]        (final output projection)
// ---------------------------------------------------------------------------
template <int MODE>
__global__ __launch_bounds__(256) void gemm_wmma(
    const _Float16* __restrict__ X,
    const _Float16* __restrict__ Wt,
    const float* __restrict__ bias,
    void* __restrict__ out) {
  const int lane = threadIdx.x & 31;
  const int lo16 = lane & 15;
  const int hi   = lane >> 4;
  const int w    = threadIdx.x >> 5;
  const int wm   = w & 3;
  const int wn   = w >> 2;
  const int bm   = blockIdx.x * 128 + wm * 32;
  const int bn   = blockIdx.y * 128 + wn * 64;

  const _Float16* xr[2];
  const _Float16* wr[4];
#pragma unroll
  for (int i = 0; i < 2; ++i) xr[i] = X  + (size_t)(bm + i * 16 + lo16) * 512;
#pragma unroll
  for (int j = 0; j < 4; ++j) wr[j] = Wt + (size_t)(bn + j * 16 + lo16) * 512;

  v8f acc[2][4] = {};

  for (int k = 0; k < 512; k += 32) {
    const int k0 = k + hi * 8;
    const int k1 = k + 16 + hi * 8;
    V16 a[2], b[4];
#pragma unroll
    for (int i = 0; i < 2; ++i) {
      a[i].h[0] = *(const v8h*)(xr[i] + k0);
      a[i].h[1] = *(const v8h*)(xr[i] + k1);
    }
#pragma unroll
    for (int j = 0; j < 4; ++j) {
      b[j].h[0] = *(const v8h*)(wr[j] + k0);
      b[j].h[1] = *(const v8h*)(wr[j] + k1);
    }
#pragma unroll
    for (int i = 0; i < 2; ++i)
#pragma unroll
      for (int j = 0; j < 4; ++j)
        acc[i][j] = __builtin_amdgcn_wmma_f32_16x16x32_f16(
            false, a[i].v, false, b[j].v, (short)0, acc[i][j], false, false);
  }

  // C layout: col = bn + j*16 + lo16 ; row = bm + i*16 + hi*8 + r
#pragma unroll
  for (int j = 0; j < 4; ++j) {
    const int gn = bn + j * 16 + lo16;
    const float bv = bias[gn];
#pragma unroll
    for (int i = 0; i < 2; ++i) {
#pragma unroll
      for (int r = 0; r < 8; ++r) {
        const int gm = bm + i * 16 + hi * 8 + r;
        const float val = acc[i][j][r] + bv;
        if (MODE == 0) {          // [B,H,L,E] f16
          const int bb = gm >> 10, l = gm & 1023, hh = gn >> 6, e = gn & 63;
          ((_Float16*)out)[(((size_t)(bb * H + hh) << 10) + l) * 64 + e] = (_Float16)val;
        } else if (MODE == 1) {   // [B,H,E,S] f16 (V transposed per head)
          const int bb = gm >> 10, s = gm & 1023, hh = gn >> 6, e = gn & 63;
          ((_Float16*)out)[(((size_t)(bb * H + hh) * 64) + e) * 1024 + s] = (_Float16)val;
        } else {                  // [M,512] f32
          ((float*)out)[(size_t)gm * 512 + gn] = val;
        }
      }
    }
  }
}

// ---------------------------------------------------------------------------
// Flash attention, one 16-row Q tile per wave, 8 waves/block; all waves in a
// block share one (b,h), so each 32-key K/V chunk is staged into LDS exactly
// once per block (TDM async tensor load, double-buffered) instead of 8x
// through the vector-memory path. Per chunk: 4 WMMAs (scores) + 4 (P@V) + 1
// (row sums via ones-column). Row max via DPP16 butterfly.
// ---------------------------------------------------------------------------
__global__ __launch_bounds__(256) void flash_attn(
    const _Float16* __restrict__ Q,
    const _Float16* __restrict__ K,
    const _Float16* __restrict__ Vt,
    _Float16* __restrict__ O16) {
  __shared__ _Float16 ldsK[2][32][64];   // double-buffered K chunk   (2x4 KB)
  __shared__ _Float16 ldsV[2][64][32];   // double-buffered V^T chunk (2x4 KB)
  __shared__ _Float16 ldsP[8][16][32];   // per-wave P repack tile    (8 KB)

  const int tid  = threadIdx.x;
  const int lane = tid & 31;
  const int lo16 = lane & 15;
  const int hi   = lane >> 4;
  const int w    = tid >> 5;
  const int bh   = blockIdx.y;           // b*H + h
  const int b    = bh >> 3;
  const int h    = bh & 7;
  const int q0   = blockIdx.x * 128 + w * 16;

  const _Float16* kbase = K  + (size_t)bh * S * E;
  const _Float16* vbase = Vt + (size_t)bh * E * S;

  // ---- stage helper: K chunk 32x64 rows, V^T chunk 64 rows x 32 cols ----
  auto stage = [&](int s0, int buf) {
#if USE_TDM
    if (w == 0) {
      tdm_load_2d(lds_off(&ldsK[buf][0][0]), kbase + (size_t)s0 * E, 64, 32, 64);
      tdm_load_2d(lds_off(&ldsV[buf][0][0]), vbase + s0, 32, 64, (unsigned)S);
    }
#else
    const int e = tid >> 2, x = (tid & 3) * 8;   // 256 threads x 16B each
    *(v8h*)(&ldsK[buf][0][0] + tid * 8) = *(const v8h*)(kbase + (size_t)s0 * E + tid * 8);
    *(v8h*)(&ldsV[buf][e][x])           = *(const v8h*)(vbase + (size_t)e * S + s0 + x);
#endif
  };

  // Q as two 16x32 A-fragments (covers E=64)
  const _Float16* qrow = Q + ((size_t)bh * L + q0 + lo16) * E;
  V16 aq0, aq1;
  aq0.h[0] = *(const v8h*)(qrow + hi * 8);
  aq0.h[1] = *(const v8h*)(qrow + 16 + hi * 8);
  aq1.h[0] = *(const v8h*)(qrow + 32 + hi * 8);
  aq1.h[1] = *(const v8h*)(qrow + 48 + hi * 8);

  // ones B-fragment: l = a*l + rowsum(P) matches the o-recurrence with a V
  // column of ones, so the matrix pipe maintains the softmax denominator.
  V16 bones;
#pragma unroll
  for (int i = 0; i < 16; ++i) bones.v[i] = (_Float16)1.0f;

  v8f o[4] = {};                          // 16 x 64 f32 accumulator
  v8f osum = {};                          // row sums (same value in all cols)
  float mrun[8];
#pragma unroll
  for (int j = 0; j < 8; ++j) mrun[j] = -1e30f;
  const float scale = 0.125f;             // 1/sqrt(E=64)

  stage(0, 0);                            // prologue: chunk 0 in flight
  int buf = 0;

  for (int s0 = 0; s0 < S; s0 += 32) {
#if USE_TDM
    __builtin_amdgcn_s_wait_tensorcnt(0); // chunk `s0` landed (wave 0)
#endif
    __syncthreads();                      // (A) chunk visible to all waves

    if (s0 + 32 < S) stage(s0 + 32, buf ^ 1);  // overlap next copy w/ compute

    // ---- scores: two 16x16 tiles over this 32-key chunk (K from LDS) ----
    v8f st[2];
#pragma unroll
    for (int t = 0; t < 2; ++t) {
      const _Float16* krow = &ldsK[buf][t * 16 + lo16][0];
      V16 bk0, bk1;
      bk0.h[0] = *(const v8h*)(krow + hi * 8);
      bk0.h[1] = *(const v8h*)(krow + 16 + hi * 8);
      bk1.h[0] = *(const v8h*)(krow + 32 + hi * 8);
      bk1.h[1] = *(const v8h*)(krow + 48 + hi * 8);
      v8f c = {};
      c = __builtin_amdgcn_wmma_f32_16x16x32_f16(false, aq0.v, false, bk0.v, (short)0, c, false, false);
      c = __builtin_amdgcn_wmma_f32_16x16x32_f16(false, aq1.v, false, bk1.v, (short)0, c, false, false);
      st[t] = c;
    }

    // ---- online softmax: DPP row-max, exp, rescale, repack P to LDS ----
#pragma unroll
    for (int j = 0; j < 8; ++j) {
      const float s0v = st[0][j] * scale;
      const float s1v = st[1][j] * scale;
      const float rmax  = rowmax16_dpp(fmaxf(s0v, s1v));
      const float newm  = fmaxf(mrun[j], rmax);
      const float alpha = __expf(mrun[j] - newm);
      const float p0 = __expf(s0v - newm);
      const float p1 = __expf(s1v - newm);
      mrun[j] = newm;
#pragma unroll
      for (int t = 0; t < 4; ++t) o[t][j] *= alpha;
      osum[j] *= alpha;
      ldsP[w][hi * 8 + j][lo16]      = (_Float16)p0;
      ldsP[w][hi * 8 + j][16 + lo16] = (_Float16)p1;
    }

    // ---- P(16x32) @ V(32x64) + ones column (V from LDS) ----
    V16 ap;
    ap.h[0] = *(const v8h*)&ldsP[w][lo16][hi * 8];
    ap.h[1] = *(const v8h*)&ldsP[w][lo16][16 + hi * 8];
#pragma unroll
    for (int t = 0; t < 4; ++t) {
      const _Float16* vrow = &ldsV[buf][t * 16 + lo16][0];
      V16 bv;
      bv.h[0] = *(const v8h*)(vrow + hi * 8);
      bv.h[1] = *(const v8h*)(vrow + 16 + hi * 8);
      o[t] = __builtin_amdgcn_wmma_f32_16x16x32_f16(false, ap.v, false, bv.v, (short)0, o[t], false, false);
    }
    osum = __builtin_amdgcn_wmma_f32_16x16x32_f16(false, ap.v, false, bones.v, (short)0, osum, false, false);

    __syncthreads();                      // (B) done reading buf this round
    buf ^= 1;
  }

  // ---- normalize and emit f16 [B*L, 512] for the output projection ----
#pragma unroll
  for (int j = 0; j < 8; ++j) {
    const float inv = 1.0f / osum[j];
    const int row = q0 + hi * 8 + j;
    _Float16* orow = O16 + ((size_t)b * L + row) * P + h * E;
#pragma unroll
    for (int t = 0; t < 4; ++t)
      orow[t * 16 + lo16] = (_Float16)(o[t][j] * inv);
  }
}

// ---------------------------------------------------------------------------
// Host-side launch
// ---------------------------------------------------------------------------
extern "C" void kernel_launch(void* const* d_in, const int* in_sizes, int n_in,
                              void* d_out, int out_size, void* d_ws, size_t ws_size,
                              hipStream_t stream) {
  const float* queries = (const float*)d_in[0];
  const float* keys    = (const float*)d_in[1];
  const float* values  = (const float*)d_in[2];
  const float* Wq = (const float*)d_in[3];
  const float* bq = (const float*)d_in[4];
  const float* Wk = (const float*)d_in[5];
  const float* bk = (const float*)d_in[6];
  const float* Wv = (const float*)d_in[7];
  const float* bv = (const float*)d_in[8];
  const float* Wo = (const float*)d_in[9];
  const float* bo = (const float*)d_in[10];
  float* out = (float*)d_out;

  char* ws = (char*)d_ws;
  size_t off = 0;
  auto take = [&](size_t bytes) -> char* {
    char* p = ws + off;
    off += (bytes + 255) & ~(size_t)255;
    return p;
  };
  const size_t actB = (size_t)MROWS * D * sizeof(_Float16);  // 16 MB
  const size_t wB   = (size_t)512 * 512 * sizeof(_Float16);  // 0.5 MB

  _Float16* Xq16 = (_Float16*)take(actB);
  _Float16* Xk16 = (_Float16*)take(actB);
  _Float16* Xv16 = (_Float16*)take(actB);
  _Float16* Wqt  = (_Float16*)take(wB);
  _Float16* Wkt  = (_Float16*)take(wB);
  _Float16* Wvt  = (_Float16*)take(wB);
  _Float16* Wot  = (_Float16*)take(wB);
  _Float16* Q16  = (_Float16*)take(actB);  // [B,H,L,64]
  _Float16* K16  = (_Float16*)take(actB);  // [B,H,S,64]
  _Float16* Vt16 = (_Float16*)take(actB);  // [B,H,64,S]
  _Float16* O16  = (_Float16*)take(actB);  // [B*L, 512]

  // 1) precision conversion (activations) and weight transpose+convert
  const int nAct = MROWS * D;                     // 8,388,608 (mult of 1024)
  cvt_f32_to_f16<<<nAct / 1024, 256, 0, stream>>>(queries, Xq16);
  cvt_f32_to_f16<<<nAct / 1024, 256, 0, stream>>>(keys,    Xk16);
  cvt_f32_to_f16<<<nAct / 1024, 256, 0, stream>>>(values,  Xv16);
  transpose_cvt_w<<<(512 * 512) / 256, 256, 0, stream>>>(Wq, Wqt);
  transpose_cvt_w<<<(512 * 512) / 256, 256, 0, stream>>>(Wk, Wkt);
  transpose_cvt_w<<<(512 * 512) / 256, 256, 0, stream>>>(Wv, Wvt);
  transpose_cvt_w<<<(512 * 512) / 256, 256, 0, stream>>>(Wo, Wot);

  // 2) QKV projections (WMMA GEMMs), block tile 128x128
  dim3 ggrid(MROWS / 128, 512 / 128);
  gemm_wmma<0><<<ggrid, 256, 0, stream>>>(Xq16, Wqt, bq, (void*)Q16);
  gemm_wmma<0><<<ggrid, 256, 0, stream>>>(Xk16, Wkt, bk, (void*)K16);
  gemm_wmma<1><<<ggrid, 256, 0, stream>>>(Xv16, Wvt, bv, (void*)Vt16);

  // 3) flash attention (online softmax, WMMA, TDM-staged K/V)
  dim3 fgrid(L / 128, B * H);
  flash_attn<<<fgrid, 256, 0, stream>>>(Q16, K16, Vt16, O16);

  // 4) output projection -> f32 d_out
  gemm_wmma<2><<<ggrid, 256, 0, stream>>>(O16, Wot, bo, (void*)out);
  (void)in_sizes; (void)n_in; (void)out_size; (void)ws_size;
}
